// GCNfusion_15564961481402
// MI455X (gfx1250) — compile-verified, matrix operand order checked by reading, not run on hardware
//
#include <hip/hip_runtime.h>
#include <hip/hip_bf16.h>

// ---------------------------------------------------------------------------
// Problem constants (fixed by the reference)
// ---------------------------------------------------------------------------
#define BB   256
#define NO   10240
#define NM   81920
#define EO   81920
#define EM   327680
#define EMB  128

typedef __attribute__((ext_vector_type(16))) _Float16 v16h;
typedef __attribute__((ext_vector_type(2)))  _Float16 h2;
typedef __attribute__((ext_vector_type(8)))  float    v8f;

// Monotone float <-> uint encoding for atomicMax on floats
__device__ __forceinline__ unsigned fenc(float x) {
    unsigned b = __float_as_uint(x);
    return (b & 0x80000000u) ? ~b : (b | 0x80000000u);
}
__device__ __forceinline__ float fdec(unsigned k) {
    return (k & 0x80000000u) ? __uint_as_float(k ^ 0x80000000u)
                             : __uint_as_float(~k);
}
#define ENC_NEG_INF 0x007FFFFFu   // fenc(-inf)

// ---------------------------------------------------------------------------
// Fills
// ---------------------------------------------------------------------------
__global__ void fill_f32_k(float* p, float v, long n) {
    long i = (long)blockIdx.x * 256 + threadIdx.x;
    if (i < n) p[i] = v;
}
__global__ void fill_u32_k(unsigned* p, unsigned v, long n) {
    long i = (long)blockIdx.x * 256 + threadIdx.x;
    if (i < n) p[i] = v;
}

// ---------------------------------------------------------------------------
// Embedding + masked mean: one block (128 threads) per node, thread = feature
// f32-out and f16-out variants (f16 feeds WMMA GEMMs directly)
// ---------------------------------------------------------------------------
__global__ void masked_mean_k(const int* __restrict__ toks,
                              const float* __restrict__ table,
                              float* __restrict__ out, int L) {
    const int node = blockIdx.x;
    const int f    = threadIdx.x;          // 0..127
    const long tb  = (long)node * L;
    float s = 0.f; int cnt = 0;
    for (int l = 0; l < L; ++l) {
        int t = toks[tb + l];
        if (t != 0) { s += table[(long)t * EMB + f]; ++cnt; }
    }
    out[(long)node * EMB + f] = s / (float)(cnt > 0 ? cnt : 1);
}
__global__ void masked_mean_h_k(const int* __restrict__ toks,
                                const float* __restrict__ table,
                                _Float16* __restrict__ out, int L) {
    const int node = blockIdx.x;
    const int f    = threadIdx.x;
    const long tb  = (long)node * L;
    float s = 0.f; int cnt = 0;
    for (int l = 0; l < L; ++l) {
        int t = toks[tb + l];
        if (t != 0) { s += table[(long)t * EMB + f]; ++cnt; }
    }
    out[(long)node * EMB + f] = (_Float16)(s / (float)(cnt > 0 ? cnt : 1));
}

// ---------------------------------------------------------------------------
// WMMA GEMM: Y[M,N] = Xh[M,K](f16) @ W[K,N](f32) (+ bias), f32 accum/out.
// Block: 256 thr (8 waves). Block tile 128(M) x 32(N), K stepped by 32.
// A tile is DMA'd with GLOBAL_LOAD_ASYNC_TO_LDS_B128 (ASYNCcnt) into
// double-buffered LDS; W tile (tiny) converted f32->f16 and ds-stored
// transposed [n][k] so all fragments are aligned 32-bit LDS reads.
// Row pad = 40 halfs (80B): 16B-aligned for async b128, stride 20 dwords is
// bank-conflict-free (20*r mod 64 distinct for r=0..15).
// M % 128 == 0, N % 32 == 0, K % 32 == 0 at every call site.
// ---------------------------------------------------------------------------
#define LDA 40
__global__ __launch_bounds__(256) void gemm_wmma_k(
    const _Float16* __restrict__ X, const float* __restrict__ W,
    const float* __restrict__ bias, float* __restrict__ Y,
    int M, int K, int N)
{
    __shared__ __attribute__((aligned(16))) _Float16 As[2][128 * LDA];
    __shared__ __attribute__((aligned(16))) _Float16 Bs[2][32  * LDA];
    const int tid  = threadIdx.x;
    const int lane = tid & 31;
    const int wave = tid >> 5;
    const int m0 = blockIdx.x * 128;
    const int n0 = blockIdx.y * 32;
    const int fm = lane & 15;               // A row / B col within 16-tile
    const int kh = (lane >> 4) << 2;        // fragment dword sub-base (0 or 4)

    // A staging: each thread DMAs two 16B chunks (rows ar, ar+64; col ac)
    const int ar = tid >> 2;                // 0..63
    const int ac = (tid & 3) * 8;           // halves, 16B-aligned chunk
    // W staging: lane-coalesced, thread owns col n, rows wk..wk+3
    const int wn = tid & 31;
    const int wk = (tid >> 5) * 4;

    // per-lane LDS byte addresses for the async DMA (low 32 bits of generic
    // shared pointer == wave-relative LDS byte offset)
    unsigned lds_a0[2], lds_a1[2];
    lds_a0[0] = (unsigned)(uintptr_t)&As[0][ar * LDA + ac];
    lds_a0[1] = (unsigned)(uintptr_t)&As[1][ar * LDA + ac];
    lds_a1[0] = (unsigned)(uintptr_t)&As[0][(ar + 64) * LDA + ac];
    lds_a1[1] = (unsigned)(uintptr_t)&As[1][(ar + 64) * LDA + ac];

    float wv[4];
    auto issue_async_A = [&](int kk, int buf) {
        unsigned long long g0 =
            (unsigned long long)(uintptr_t)(X + (size_t)(m0 + ar) * K + kk + ac);
        unsigned long long g1 =
            (unsigned long long)(uintptr_t)(X + (size_t)(m0 + ar + 64) * K + kk + ac);
        asm volatile("global_load_async_to_lds_b128 %0, %1, off"
                     :: "v"(lds_a0[buf]), "v"(g0) : "memory");
        asm volatile("global_load_async_to_lds_b128 %0, %1, off"
                     :: "v"(lds_a1[buf]), "v"(g1) : "memory");
    };
    auto load_W = [&](int kk) {
        #pragma unroll
        for (int i = 0; i < 4; ++i)
            wv[i] = W[(size_t)(kk + wk + i) * N + n0 + wn];   // coalesced
    };
    auto store_W = [&](int buf) {
        h2* dp = (h2*)&Bs[buf][wn * LDA + wk];
        h2 t0; t0.x = (_Float16)wv[0]; t0.y = (_Float16)wv[1];
        h2 t1; t1.x = (_Float16)wv[2]; t1.y = (_Float16)wv[3];
        dp[0] = t0; dp[1] = t1;
    };

    v8f acc0 = {0.f,0.f,0.f,0.f,0.f,0.f,0.f,0.f};
    v8f acc1 = acc0;

    // prologue: fill buffer 0
    issue_async_A(0, 0);
    load_W(0);
    store_W(0);
    asm volatile("s_wait_asynccnt 0" ::: "memory");
    __syncthreads();

    const int nsteps = K >> 5;
    int buf = 0;
    const int ab  = (wave * 16 + fm) * (LDA / 2);  // dword row base in As
    const int bb0 = fm * (LDA / 2);                // dword row base in Bs (n)
    const int bb1 = (fm + 16) * (LDA / 2);

    for (int s = 0; s < nsteps; ++s) {
        if (s + 1 < nsteps) {                       // prefetch next tile
            issue_async_A((s + 1) << 5, buf ^ 1);   // DMA A (overlaps WMMAs)
            load_W((s + 1) << 5);
        }

        union U { v16h h; unsigned u[8]; } a, b0, b1;
        const unsigned* A32 = (const unsigned*)As[buf];
        const unsigned* B32 = (const unsigned*)Bs[buf];
        #pragma unroll
        for (int j = 0; j < 8; ++j) {
            int ko = (j < 4) ? (kh + j) : (kh + 4 + j);  // packed K-pair dword
            a.u[j]  = A32[ab  + ko];
            b0.u[j] = B32[bb0 + ko];
            b1.u[j] = B32[bb1 + ko];
        }
        acc0 = __builtin_amdgcn_wmma_f32_16x16x32_f16(false, a.h, false, b0.h,
                                                      (short)0, acc0, false, false);
        acc1 = __builtin_amdgcn_wmma_f32_16x16x32_f16(false, a.h, false, b1.h,
                                                      (short)0, acc1, false, false);

        if (s + 1 < nsteps) {
            store_W(buf ^ 1);
            asm volatile("s_wait_asynccnt 0" ::: "memory"); // DMA done
            __syncthreads();                                 // all stores visible
        }
        buf ^= 1;
    }

    // store per C/D f32 layout: VGPR i -> M = i + (lane<16 ? 0 : 8), N = lane&15
    const int nc = fm;
    const int mh = (lane >> 4) << 3;
    const float bv0 = bias ? bias[n0 + nc]      : 0.f;
    const float bv1 = bias ? bias[n0 + 16 + nc] : 0.f;
    #pragma unroll
    for (int i = 0; i < 8; ++i) {
        int row = m0 + wave * 16 + mh + i;
        Y[(size_t)row * N + n0 + nc]      = acc0[i] + bv0;
        Y[(size_t)row * N + n0 + 16 + nc] = acc1[i] + bv1;
    }
}

// ---------------------------------------------------------------------------
// TransformerConv edge passes (H=8 heads, Dh=16)
// ---------------------------------------------------------------------------
__global__ void edge_score_k(const float* __restrict__ q, const float* __restrict__ k,
                             const int* __restrict__ src, const int* __restrict__ dst,
                             float* __restrict__ score, unsigned* __restrict__ smax) {
    long idx = (long)blockIdx.x * 256 + threadIdx.x;
    if (idx >= (long)EM * 8) return;
    long e = idx >> 3; int h = (int)(idx & 7);
    int d = dst[e], s = src[e];
    const float4* qp = (const float4*)(q + (long)d * EMB + h * 16);
    const float4* kp = (const float4*)(k + (long)s * EMB + h * 16);
    float acc = 0.f;
    #pragma unroll
    for (int i = 0; i < 4; ++i) {
        float4 a = qp[i], b = kp[i];
        acc += a.x * b.x + a.y * b.y + a.z * b.z + a.w * b.w;
    }
    acc *= 0.25f;                            // 1/sqrt(Dh)
    score[idx] = acc;
    atomicMax(&smax[(long)d * 8 + h], fenc(acc));
}

__global__ void edge_exp_k(float* __restrict__ score,
                           const unsigned* __restrict__ smax,
                           float* __restrict__ ssum,
                           const int* __restrict__ dst) {
    long idx = (long)blockIdx.x * 256 + threadIdx.x;
    if (idx >= (long)EM * 8) return;
    long e = idx >> 3; int h = (int)(idx & 7);
    int d = dst[e];
    float ev = __expf(score[idx] - fdec(smax[(long)d * 8 + h]));
    score[idx] = ev;
    atomicAdd(&ssum[(long)d * 8 + h], ev);
}

__global__ void edge_agg_k(const float* __restrict__ score,
                           const float* __restrict__ ssum,
                           const float* __restrict__ v,
                           const int* __restrict__ src, const int* __restrict__ dst,
                           float* __restrict__ hid) {
    long idx = (long)blockIdx.x * 256 + threadIdx.x;
    if (idx >= (long)EM * EMB) return;
    long e = idx >> 7; int c = (int)(idx & 127); int h = c >> 4;
    int d = dst[e], s = src[e];
    float alpha = score[e * 8 + h] / (ssum[(long)d * 8 + h] + 1e-16f);
    atomicAdd(&hid[(long)d * EMB + c], alpha * v[(long)s * EMB + c]);
}

// ---------------------------------------------------------------------------
// Global attention helpers
// ---------------------------------------------------------------------------
__global__ void rowdot_k(const float* __restrict__ x, const float* __restrict__ wg,
                         const float* __restrict__ bg, float* __restrict__ gate, int n) {
    int node = blockIdx.x * 8 + (threadIdx.x >> 5);
    int lane = threadIdx.x & 31;
    if (node >= n) return;
    float s = 0.f;
    #pragma unroll
    for (int i = 0; i < 4; ++i)
        s += x[(long)node * EMB + lane + 32 * i] * wg[lane + 32 * i];
    #pragma unroll
    for (int off = 16; off; off >>= 1) s += __shfl_xor(s, off, 32);
    if (lane == 0) gate[node] = s + bg[0];
}

__global__ void seg_max_k(const float* __restrict__ gate, const int* __restrict__ seg,
                          unsigned* __restrict__ keys, int n) {
    int i = blockIdx.x * 256 + threadIdx.x;
    if (i < n) atomicMax(&keys[seg[i]], fenc(gate[i]));
}
__global__ void seg_exp_k(float* __restrict__ gate, const unsigned* __restrict__ keys,
                          float* __restrict__ sums, const int* __restrict__ seg, int n) {
    int i = blockIdx.x * 256 + threadIdx.x;
    if (i < n) {
        float e = __expf(gate[i] - fdec(keys[seg[i]]));
        gate[i] = e;
        atomicAdd(&sums[seg[i]], e);
    }
}
__global__ void seg_agg_k(const float* __restrict__ gate, const float* __restrict__ sums,
                          const float* __restrict__ x, const int* __restrict__ seg,
                          float* __restrict__ out, long n) {
    long idx = (long)blockIdx.x * 256 + threadIdx.x;
    if (idx >= n * EMB) return;
    long i = idx >> 7; int f = (int)(idx & 127);
    int sg = seg[i];
    float a = gate[i] / (sums[sg] + 1e-16f);
    atomicAdd(&out[(long)sg * EMB + f], a * x[idx]);
}

// ---------------------------------------------------------------------------
// Misc elementwise + GCN helpers
// ---------------------------------------------------------------------------
// mini_fn = (ga_out + stmt)*0.5, written as f16 (feeds GEMM only)
__global__ void combine_h_k(const float* __restrict__ a, const float* __restrict__ b,
                            _Float16* __restrict__ out, long n) {
    long i = (long)blockIdx.x * 256 + threadIdx.x;
    if (i < n) out[i] = (_Float16)((a[i] + b[i]) * 0.5f);
}
// h = relu(x), written as f16 (feeds GEMM only)
__global__ void relu_h_k(const float* __restrict__ x, _Float16* __restrict__ out, long n) {
    long i = (long)blockIdx.x * 256 + threadIdx.x;
    if (i < n) out[i] = (_Float16)fmaxf(x[i], 0.f);
}
__global__ void deg_edge_k(const int* __restrict__ dst, float* __restrict__ deg, long n) {
    long i = (long)blockIdx.x * 256 + threadIdx.x;
    if (i < n) atomicAdd(&deg[dst[i]], 1.f);
}
__global__ void rsqrt_k(float* p, int n) {
    int i = blockIdx.x * 256 + threadIdx.x;
    if (i < n) p[i] = rsqrtf(p[i]);
}
// out = b[f] + xw * dinv[i]^2   (bias + self-loop term)
__global__ void gcn_init_k(float* __restrict__ out, const float* __restrict__ xw,
                           const float* __restrict__ dinv, const float* __restrict__ b,
                           long n, int fshift) {
    long idx = (long)blockIdx.x * 256 + threadIdx.x;
    if (idx >= n) return;
    long i = idx >> fshift; int f = (int)(idx & ((1 << fshift) - 1));
    float di = dinv[i];
    out[idx] = b[f] + xw[idx] * di * di;
}
__global__ void gcn_scatter_k(float* __restrict__ out, const float* __restrict__ xw,
                              const float* __restrict__ dinv,
                              const int* __restrict__ src, const int* __restrict__ dst,
                              long n, int fshift) {
    long idx = (long)blockIdx.x * 256 + threadIdx.x;
    if (idx >= n) return;
    long e = idx >> fshift; int f = (int)(idx & ((1 << fshift) - 1));
    int s = src[e], d = dst[e];
    atomicAdd(&out[((long)d << fshift) + f],
              xw[((long)s << fshift) + f] * dinv[s] * dinv[d]);
}

// ---------------------------------------------------------------------------
// Cosine similarity: one wave per batch row
// ---------------------------------------------------------------------------
__global__ void cosine_k(const float* __restrict__ fr, const float* __restrict__ hn,
                         float* __restrict__ out) {
    int b = blockIdx.x * 8 + (threadIdx.x >> 5);
    int lane = threadIdx.x & 31;
    float dot = 0.f, na = 0.f, nb = 0.f;
    #pragma unroll
    for (int i = 0; i < 4; ++i) {
        float a = fr[(long)b * EMB + lane + 32 * i];
        float c = hn[(long)b * EMB + lane + 32 * i];
        dot += a * c; na += a * a; nb += c * c;
    }
    #pragma unroll
    for (int off = 16; off; off >>= 1) {
        dot += __shfl_xor(dot, off, 32);
        na  += __shfl_xor(na,  off, 32);
        nb  += __shfl_xor(nb,  off, 32);
    }
    if (lane == 0)
        out[b] = dot / (fmaxf(sqrtf(na), 1e-8f) * fmaxf(sqrtf(nb), 1e-8f));
}

// ---------------------------------------------------------------------------
// Host-side launcher
// ---------------------------------------------------------------------------
static inline unsigned gblk(long n) { return (unsigned)((n + 255) / 256); }

extern "C" void kernel_launch(void* const* d_in, const int* in_sizes, int n_in,
                              void* d_out, int out_size, void* d_ws, size_t ws_size,
                              hipStream_t stream) {
    (void)in_sizes; (void)n_in; (void)out_size; (void)ws_size;

    const int*   desc_tokens = (const int*)  d_in[0];
    const int*   x_tokens    = (const int*)  d_in[1];
    const int*   mini_tokens = (const int*)  d_in[2];
    const int*   src         = (const int*)  d_in[3];
    const int*   dst         = (const int*)  d_in[4];
    const int*   mini_src    = (const int*)  d_in[5];
    const int*   mini_dst    = (const int*)  d_in[6];
    const int*   mini_batch  = (const int*)  d_in[7];
    const int*   node_batch  = (const int*)  d_in[8];
    const float* desc_table  = (const float*)d_in[9];
    const float* code_table  = (const float*)d_in[10];
    const float* code_table2 = (const float*)d_in[11];
    const float* Wq = (const float*)d_in[12]; const float* bq = (const float*)d_in[13];
    const float* Wk = (const float*)d_in[14]; const float* bk = (const float*)d_in[15];
    const float* Wv = (const float*)d_in[16]; const float* bv = (const float*)d_in[17];
    const float* Wsk= (const float*)d_in[18]; const float* bsk= (const float*)d_in[19];
    const float* W2 = (const float*)d_in[20]; const float* b2 = (const float*)d_in[21];
    const float* W3 = (const float*)d_in[22]; const float* b3 = (const float*)d_in[23];
    const float* Wg = (const float*)d_in[24]; const float* bg = (const float*)d_in[25];

    float* ws = (float*)d_ws;
    const size_t NMF = (size_t)NM * EMB, NOF = (size_t)NO * EMB;
    size_t o_mini16 = 0;                            // NM*128 f16 (NMF/2 floats)
    size_t o_q      = o_mini16 + NMF / 2;
    size_t o_k      = o_q      + NMF;
    size_t o_v      = o_k      + NMF;
    size_t o_hid    = o_v      + NMF;               // skip GEMM -> agg -> mini_hidden
    size_t o_score  = o_hid    + NMF;               // EM*8
    size_t o_smax   = o_score  + (size_t)EM * 8;    // NM*8
    size_t o_ssum   = o_smax   + (size_t)NM * 8;    // NM*8
    size_t o_stmt   = o_ssum   + (size_t)NM * 8;    // NO*128
    size_t o_mfn    = o_stmt   + NOF;               // NO*128 (f32 accum)
    size_t o_mfn16  = o_mfn    + NOF;               // NO*128 f16
    size_t o_gate   = o_mfn16  + NOF / 2;           // NM
    size_t o_gmax   = o_gate   + NM;                // NO (>= B)
    size_t o_gsum   = o_gmax   + NO;                // NO
    size_t o_dinv   = o_gsum   + NO;                // NO
    size_t o_xw1    = o_dinv   + NO;                // NO*256
    size_t o_h      = o_xw1    + (size_t)NO * 256;  // NO*256 (f32 accum)
    size_t o_h16    = o_h      + (size_t)NO * 256;  // NO*256 f16
    size_t o_xw2    = o_h16    + (size_t)NO * 128;  // NO*128
    size_t o_fin    = o_xw2    + NOF;               // NO*128
    size_t o_fnr    = o_fin    + NOF;               // B*128
    size_t o_hn     = o_fnr    + (size_t)BB * EMB;  // B*128

    _Float16* p_mini16 = (_Float16*)(ws + o_mini16);
    _Float16* p_mfn16  = (_Float16*)(ws + o_mfn16);
    _Float16* p_h16    = (_Float16*)(ws + o_h16);
    float* p_q   = ws + o_q;   float* p_kk  = ws + o_k;   float* p_v   = ws + o_v;
    float* p_hid = ws + o_hid; float* p_sc  = ws + o_score;
    float* p_stmt= ws + o_stmt;float* p_mfn = ws + o_mfn; float* p_gate= ws + o_gate;
    float* p_dinv= ws + o_dinv;float* p_xw1 = ws + o_xw1; float* p_h   = ws + o_h;
    float* p_xw2 = ws + o_xw2; float* p_fin = ws + o_fin; float* p_fnr = ws + o_fnr;
    float* p_hn  = ws + o_hn;
    unsigned* p_smax = (unsigned*)(ws + o_smax); float* p_ssum = ws + o_ssum;
    unsigned* p_gmax = (unsigned*)(ws + o_gmax); float* p_gsum = ws + o_gsum;

    // 1) embeddings + masked mean (mini_emb straight to f16: GEMM-only consumer)
    masked_mean_k  <<<BB, EMB, 0, stream>>>(desc_tokens, desc_table,  p_hn,     64);
    masked_mean_k  <<<NO, EMB, 0, stream>>>(x_tokens,    code_table2, p_stmt,   32);
    masked_mean_h_k<<<NM, EMB, 0, stream>>>(mini_tokens, code_table,  p_mini16, 16);

    // 2) TransformerConv dense projections (WMMA f16, async-LDS A tiles)
    gemm_wmma_k<<<dim3(NM/128, 128/32), 256, 0, stream>>>(p_mini16, Wq,  bq,  p_q,   NM, 128, 128);
    gemm_wmma_k<<<dim3(NM/128, 128/32), 256, 0, stream>>>(p_mini16, Wk,  bk,  p_kk,  NM, 128, 128);
    gemm_wmma_k<<<dim3(NM/128, 128/32), 256, 0, stream>>>(p_mini16, Wv,  bv,  p_v,   NM, 128, 128);
    gemm_wmma_k<<<dim3(NM/128, 128/32), 256, 0, stream>>>(p_mini16, Wsk, bsk, p_hid, NM, 128, 128);

    // 3) edge attention: segment softmax over dst, aggregate into p_hid (skip-init)
    fill_u32_k<<<gblk((long)NM*8), 256, 0, stream>>>(p_smax, ENC_NEG_INF, (long)NM*8);
    fill_f32_k<<<gblk((long)NM*8), 256, 0, stream>>>(p_ssum, 0.f, (long)NM*8);
    edge_score_k<<<gblk((long)EM*8),   256, 0, stream>>>(p_q, p_kk, mini_src, mini_dst, p_sc, p_smax);
    edge_exp_k  <<<gblk((long)EM*8),   256, 0, stream>>>(p_sc, p_smax, p_ssum, mini_dst);
    edge_agg_k  <<<gblk((long)EM*EMB), 256, 0, stream>>>(p_sc, p_ssum, p_v, mini_src, mini_dst, p_hid);

    // 4) global attention #1: mini_hidden -> mini_fn over mini_batch (NO segments)
    rowdot_k<<<NM/8, 256, 0, stream>>>(p_hid, Wg, bg, p_gate, NM);
    fill_u32_k<<<gblk(NO), 256, 0, stream>>>(p_gmax, ENC_NEG_INF, NO);
    fill_f32_k<<<gblk(NO), 256, 0, stream>>>(p_gsum, 0.f, NO);
    fill_f32_k<<<gblk((long)NO*EMB), 256, 0, stream>>>(p_mfn, 0.f, (long)NO*EMB);
    seg_max_k<<<gblk(NM), 256, 0, stream>>>(p_gate, mini_batch, p_gmax, NM);
    seg_exp_k<<<gblk(NM), 256, 0, stream>>>(p_gate, p_gmax, p_gsum, mini_batch, NM);
    seg_agg_k<<<gblk((long)NM*EMB), 256, 0, stream>>>(p_gate, p_gsum, p_hid, mini_batch, p_mfn, NM);
    combine_h_k<<<gblk((long)NO*EMB), 256, 0, stream>>>(p_mfn, p_stmt, p_mfn16, (long)NO*EMB);

    // 5) GCN degree normalization (shared by both convs); deg = 1 + indegree
    fill_f32_k<<<gblk(NO), 256, 0, stream>>>(p_dinv, 1.f, NO);
    deg_edge_k<<<gblk(EO), 256, 0, stream>>>(dst, p_dinv, EO);
    rsqrt_k<<<gblk(NO), 256, 0, stream>>>(p_dinv, NO);

    // 6) GCN layer 1: h = relu(scatter(mini_fn @ W2) + b2)  (h -> f16 for GEMM2)
    gemm_wmma_k<<<dim3(NO/128, 256/32), 256, 0, stream>>>(p_mfn16, W2, nullptr, p_xw1, NO, 128, 256);
    gcn_init_k   <<<gblk((long)NO*256), 256, 0, stream>>>(p_h, p_xw1, p_dinv, b2, (long)NO*256, 8);
    gcn_scatter_k<<<gblk((long)EO*256), 256, 0, stream>>>(p_h, p_xw1, p_dinv, src, dst, (long)EO*256, 8);
    relu_h_k<<<gblk((long)NO*256), 256, 0, stream>>>(p_h, p_h16, (long)NO*256);

    // 7) GCN layer 2: final_stmt = scatter(h @ W3) + b3
    gemm_wmma_k<<<dim3(NO/128, 128/32), 256, 0, stream>>>(p_h16, W3, nullptr, p_xw2, NO, 256, 128);
    gcn_init_k   <<<gblk((long)NO*EMB), 256, 0, stream>>>(p_fin, p_xw2, p_dinv, b3, (long)NO*EMB, 7);
    gcn_scatter_k<<<gblk((long)EO*EMB), 256, 0, stream>>>(p_fin, p_xw2, p_dinv, src, dst, (long)EO*EMB, 7);

    // 8) global attention #2: final_stmt -> fn_repr over node_batch (B segments)
    rowdot_k<<<NO/8, 256, 0, stream>>>(p_fin, Wg, bg, p_gate, NO);
    fill_u32_k<<<gblk(BB), 256, 0, stream>>>(p_gmax, ENC_NEG_INF, BB);
    fill_f32_k<<<gblk(BB), 256, 0, stream>>>(p_gsum, 0.f, BB);
    fill_f32_k<<<gblk((long)BB*EMB), 256, 0, stream>>>(p_fnr, 0.f, (long)BB*EMB);
    seg_max_k<<<gblk(NO), 256, 0, stream>>>(p_gate, node_batch, p_gmax, NO);
    seg_exp_k<<<gblk(NO), 256, 0, stream>>>(p_gate, p_gmax, p_gsum, node_batch, NO);
    seg_agg_k<<<gblk((long)NO*EMB), 256, 0, stream>>>(p_gate, p_gsum, p_fin, node_batch, p_fnr, NO);

    // 9) cosine similarity -> d_out[256]
    cosine_k<<<BB/8, 256, 0, stream>>>(p_fnr, p_hn, (float*)d_out);
}
